// CategoryKernel_89764816486452
// MI455X (gfx1250) — 8-bit matrix operand order repaired and checked by reading, compiled, not run on hardware
//
#include <hip/hip_runtime.h>


typedef __attribute__((ext_vector_type(8))) int   v8i;
typedef __attribute__((ext_vector_type(8))) float v8f;

#define COLS_PER_WAVE   8
#define WAVES_PER_BLOCK 8

#define FP8_ONE 0x38u

__device__ inline v8i onehot_A(int z, int h) {
    const int hreq = (z >> 4) & 1;
    const int vg   = ((z >> 5) << 2) | ((z >> 2) & 3);
    const unsigned word = FP8_ONE << ((z & 3) * 8);
    v8i a;
#pragma unroll
    for (int i = 0; i < 8; ++i)
        a[i] = ((hreq == h) && (vg == i)) ? (int)word : 0;
    return a;
}

__device__ inline v8i onehot_B(int z, int h) {
    const int hreq = (z >> 4) & 1;
    const int vg   = ((z >> 5) << 2) | ((z >> 2) & 3);
    const unsigned word = FP8_ONE << ((z & 3) * 8);
    v8i b;
#pragma unroll
    for (int i = 0; i < 8; ++i)
        b[i] = ((hreq == h) && (vg == i)) ? (int)word : 0;
    return b;
}

__global__ __launch_bounds__(256) void category_gram_wmma(
    const int* __restrict__ Z, float* __restrict__ out, int N)
{
    const int lane = threadIdx.x;
    const int wave = threadIdx.y;
    const int h    = lane >> 4;
    const int m    = lane & 15;

    const int row_base  = blockIdx.x * 16;
    const int col_group = blockIdx.y * WAVES_PER_BLOCK + wave;
    const int col_base0 = col_group * (COLS_PER_WAVE * 16);

    const int zr = Z[row_base + m];
    const v8i a = onehot_A(zr, h);

#pragma unroll
    for (int t = 0; t < COLS_PER_WAVE; ++t) {
        const int col_base = col_base0 + t * 16;

        const int zc = Z[col_base + m];
        const v8i b = onehot_B(zc, h);

        v8f c = {};
        c = __builtin_amdgcn_wmma_f32_16x16x64_fp8_fp8(
                a, b,  (short)0, c,  false,  false);

        float* p = out + (size_t)(row_base + 8 * h) * (size_t)N
                       + (size_t)(col_base + m);
#pragma unroll
        for (int r = 0; r < 8; ++r)
            __builtin_nontemporal_store(c[r], p + (size_t)r * (size_t)N);
    }
}

extern "C" void kernel_launch(void* const* d_in, const int* in_sizes, int n_in,
                              void* d_out, int out_size, void* d_ws, size_t ws_size,
                              hipStream_t stream) {
    const int* Z   = (const int*)d_in[0];
    float*     out = (float*)d_out;
    const int  N   = in_sizes[0];
    const int  NT  = N / 16;

    dim3 block(32, WAVES_PER_BLOCK);
    dim3 grid(NT, NT / (COLS_PER_WAVE * WAVES_PER_BLOCK));
    hipLaunchKernelGGL(category_gram_wmma, grid, block, 0, stream, Z, out, N);
}
